// LongformerAttention_77232101917207
// MI455X (gfx1250) — matrix operand order repaired and checked by reading, not compiled, not run on hardware
//
#include <hip/hip_runtime.h>
#include <hip/hip_bf16.h>
#include <math.h>

// Problem constants (match reference)
#define B_   2
#define S_   2048
#define D_   1024
#define H_   16
#define DH_  64
#define WIN_ 128

typedef __attribute__((ext_vector_type(16))) __bf16 v16bf;
typedef __attribute__((ext_vector_type(8)))  __bf16 v8bf;
typedef __attribute__((ext_vector_type(8)))  float  v8f;

// ---------------------------------------------------------------------------
// WMMA helper:  D = A(16x32 bf16) * B(32x16 bf16) + C(16x16 f32)
// ---------------------------------------------------------------------------
__device__ __forceinline__ v8f wmma_bf16(v16bf a, v16bf b, v8f c) {
  return __builtin_amdgcn_wmma_f32_16x16x32_bf16(
      /*neg_a=*/false, a, /*neg_b=*/false, b,
      /*c_mod=*/(short)0, c, /*reuse_a=*/false, /*reuse_b=*/false);
}

// A-fragment (16x32, bf16) from a row-major matrix row.
// Layout: lanes 0-15 hold row (lane&15), K = kb+0..15;
//         lanes 16-31 hold same rows, K = kb+16..31.
__device__ __forceinline__ v16bf load_a_frag(const __bf16* row, int kb, int lane) {
  const int off = (lane >= 16) ? 16 : 0;
  v8bf lo = *reinterpret_cast<const v8bf*>(row + kb + off);
  v8bf hi = *reinterpret_cast<const v8bf*>(row + kb + 8 + off);
  return __builtin_shufflevector(lo, hi, 0, 1, 2, 3, 4, 5, 6, 7,
                                        8, 9, 10, 11, 12, 13, 14, 15);
}

// B-fragment (32x16, bf16) where B[k][n] = Wrow_n[k] (row of W = column of W^T).
// Lane holds column n = lane&15; 16 consecutive K values per lane:
// lanes 0-15 -> K = kb+0..15, lanes 16-31 -> K = kb+16..31.  One 32B load.
__device__ __forceinline__ v16bf load_b_frag_row(const __bf16* wrow, int kb, int lane) {
  const int off = (lane >= 16) ? 16 : 0;
  return *reinterpret_cast<const v16bf*>(wrow + kb + off);
}

// ---------------------------------------------------------------------------
// fp32 -> bf16 convert
// ---------------------------------------------------------------------------
__global__ void cvt_f32_bf16(const float* __restrict__ src,
                             __bf16* __restrict__ dst, int n) {
  int i = blockIdx.x * blockDim.x + threadIdx.x;
  if (i < n) dst[i] = (__bf16)src[i];
}

// ---------------------------------------------------------------------------
// attn region zero fill (float4 stores)
// ---------------------------------------------------------------------------
__global__ void zero_f4(float4* __restrict__ p, size_t n4) {
  size_t i = (size_t)blockIdx.x * blockDim.x + threadIdx.x;
  if (i < n4) p[i] = make_float4(0.f, 0.f, 0.f, 0.f);
}

// ---------------------------------------------------------------------------
// Fused QKV projection:  Y = X @ W^T + b,  stored bf16 in [B,H,S,dh] layout.
// Register-blocked 2x4: each wave owns a 32x64 output tile (8 acc tiles).
// Per K=32 step: 2 A-frags + 4 B-frags (12 b128 loads) feed 8 WMMAs.
// grid = (M/32/8, D/64, 3), block = 256 (8 waves)
// ---------------------------------------------------------------------------
#define MB_ 2
#define NB_ 4
__global__ __launch_bounds__(256) void qkv_gemm(
    const __bf16* __restrict__ xb,
    const __bf16* __restrict__ wq, const __bf16* __restrict__ wk,
    const __bf16* __restrict__ wv,
    const float* __restrict__ bq, const float* __restrict__ bk,
    const float* __restrict__ bv,
    __bf16* __restrict__ qo, __bf16* __restrict__ ko, __bf16* __restrict__ vo) {
  const int lane  = threadIdx.x & 31;
  const int wid   = threadIdx.x >> 5;
  const int mt    = blockIdx.x * 8 + wid;   // 0..127  (M = 4096, 32 rows/wave)
  const int nt    = blockIdx.y;             // 0..15   (N = 1024, 64 cols/wave)
  const int which = blockIdx.z;

  const __bf16* w    = (which == 0) ? wq : (which == 1) ? wk : wv;
  const float*  bias = (which == 0) ? bq : (which == 1) ? bk : bv;
  __bf16*       out  = (which == 0) ? qo : (which == 1) ? ko : vo;

  const int m0 = mt * (16 * MB_), n0 = nt * (16 * NB_);
  const int nloc = lane & 15;

  const __bf16* arow[MB_];
#pragma unroll
  for (int mi = 0; mi < MB_; ++mi)
    arow[mi] = xb + (size_t)(m0 + mi * 16 + nloc) * D_;
  const __bf16* brow[NB_];
#pragma unroll
  for (int ni = 0; ni < NB_; ++ni)
    brow[ni] = w + (size_t)(n0 + ni * 16 + nloc) * D_;

  v8f acc[MB_][NB_];
#pragma unroll
  for (int mi = 0; mi < MB_; ++mi)
#pragma unroll
    for (int ni = 0; ni < NB_; ++ni) acc[mi][ni] = (v8f){};

#pragma unroll 2
  for (int kb = 0; kb < D_; kb += 32) {
    v16bf a[MB_], b[NB_];
#pragma unroll
    for (int mi = 0; mi < MB_; ++mi) a[mi] = load_a_frag(arow[mi], kb, lane);
#pragma unroll
    for (int ni = 0; ni < NB_; ++ni) b[ni] = load_b_frag_row(brow[ni], kb, lane);
#pragma unroll
    for (int mi = 0; mi < MB_; ++mi)
#pragma unroll
      for (int ni = 0; ni < NB_; ++ni)
        acc[mi][ni] = wmma_bf16(a[mi], b[ni], acc[mi][ni]);
  }

  const int half8 = (lane >= 16) ? 8 : 0;
#pragma unroll
  for (int ni = 0; ni < NB_; ++ni) {
    const int ncol = n0 + ni * 16 + nloc;
    const float bval = bias[ncol];
    const int h = ncol >> 6;   // head
    const int e = ncol & 63;   // dim within head
#pragma unroll
    for (int mi = 0; mi < MB_; ++mi) {
#pragma unroll
      for (int r = 0; r < 8; ++r) {
        const int m  = m0 + mi * 16 + r + half8;
        const int bb = m / S_;
        const int s  = m % S_;
        out[(((size_t)bb * H_ + h) * S_ + s) * DH_ + e] =
            (__bf16)(acc[mi][ni][r] + bval);
      }
    }
  }
}

// ---------------------------------------------------------------------------
// Band attention.  One wave = 16 queries of one (b,h).
// Pass 1: row max & exp-sum via WMMA scores + shfl reductions.
// Pass 2: recompute scores, write normalized probs to attn, P*V via WMMA
//         with P transposed through LDS (C-layout -> A-layout).
// grid = (S/16/4, H, B), block = 128 (4 independent waves)
// ---------------------------------------------------------------------------
__global__ __launch_bounds__(128) void attn_kernel(
    const __bf16* __restrict__ qb, const __bf16* __restrict__ kb,
    const __bf16* __restrict__ vb, __bf16* __restrict__ ctxb,
    float* __restrict__ attn) {
  __shared__ __bf16 pstage[4][16 * 32];   // per-wave P tile (16 q x 32 k)

  const int lane = threadIdx.x & 31;
  const int wid  = threadIdx.x >> 5;
  const int qt   = blockIdx.x * 4 + wid;  // 0..127
  const int h    = blockIdx.y;
  const int bb   = blockIdx.z;
  const int q0   = qt * 16;

  const size_t headoff = ((size_t)bb * H_ + h) * S_ * DH_;
  const __bf16* qbase = qb + headoff;
  const __bf16* kbase = kb + headoff;
  const __bf16* vbase = vb + headoff;

  const int nloc  = lane & 15;
  const int half8 = (lane >= 16) ? 8 : 0;

  // Q fragments (row = q0 + (lane&15)), dh = 64 -> two K=32 fragments
  const __bf16* qrow = qbase + (size_t)(q0 + nloc) * DH_;
  const v16bf aq0 = load_a_frag(qrow, 0, lane);
  const v16bf aq1 = load_a_frag(qrow, 32, lane);

  const int klo    = (q0 - WIN_ > 0) ? (q0 - WIN_) : 0;
  const int khi    = (q0 + 16 + WIN_ < S_) ? (q0 + 16 + WIN_) : S_;
  const int ntiles = (khi - klo) >> 4;
  const float scale = 0.125f;  // 1/sqrt(64)

  float m_[8], l_[8];
#pragma unroll
  for (int r = 0; r < 8; ++r) { m_[r] = -__builtin_inff(); l_[r] = 0.f; }

  // ---- pass 1: softmax statistics -----------------------------------------
  for (int t = 0; t < ntiles; ++t) {
    const int k0 = klo + t * 16;
    const __bf16* krow = kbase + (size_t)(k0 + nloc) * DH_;
    if (t + 1 < ntiles) __builtin_prefetch(krow + 16 * DH_, 0, 0);
    v16bf bk0 = load_b_frag_row(krow, 0, lane);
    v16bf bk1 = load_b_frag_row(krow, 32, lane);
    v8f c = {};
    c = wmma_bf16(aq0, bk0, c);
    c = wmma_bf16(aq1, bk1, c);
    const int j = k0 + nloc;
#pragma unroll
    for (int r = 0; r < 8; ++r) {
      const int i = q0 + r + half8;
      float val = c[r] * scale;
      const bool ok = (j >= i - WIN_) && (j <= i + WIN_);
      val = ok ? val : -__builtin_inff();
      float vmax = val;
#pragma unroll
      for (int sh = 8; sh >= 1; sh >>= 1)
        vmax = fmaxf(vmax, __shfl_xor(vmax, sh, 32));
      const float mnew = fmaxf(m_[r], vmax);
      float p = __expf(val - mnew);
#pragma unroll
      for (int sh = 8; sh >= 1; sh >>= 1) p += __shfl_xor(p, sh, 32);
      l_[r] = l_[r] * __expf(m_[r] - mnew) + p;
      m_[r] = mnew;
    }
  }

  float invl[8];
#pragma unroll
  for (int r = 0; r < 8; ++r) invl[r] = 1.f / l_[r];

  // ---- pass 2: probabilities out + P*V ------------------------------------
  v8f acc[4] = {{}, {}, {}, {}};   // ctx tiles: 16 x (4*16) dims
  __bf16* myp = pstage[wid];
  const int npairs = (ntiles + 1) >> 1;

  for (int pp = 0; pp < npairs; ++pp) {
#pragma unroll
    for (int sub = 0; sub < 2; ++sub) {
      const int t = pp * 2 + sub;
      if (t < ntiles) {
        const int k0 = klo + t * 16;
        const __bf16* krow = kbase + (size_t)(k0 + nloc) * DH_;
        v16bf bk0 = load_b_frag_row(krow, 0, lane);
        v16bf bk1 = load_b_frag_row(krow, 32, lane);
        v8f c = {};
        c = wmma_bf16(aq0, bk0, c);
        c = wmma_bf16(aq1, bk1, c);
        const int j = k0 + nloc;
#pragma unroll
        for (int r = 0; r < 8; ++r) {
          const int i = q0 + r + half8;
          const bool ok = (j >= i - WIN_) && (j <= i + WIN_);
          const float p = ok ? __expf(c[r] * scale - m_[r]) * invl[r] : 0.f;
          attn[(((size_t)bb * H_ + h) * S_ + i) * S_ + j] = p;
          myp[(r + half8) * 32 + sub * 16 + nloc] = (__bf16)p;
        }
      } else {
#pragma unroll
        for (int r = 0; r < 8; ++r)
          myp[(r + half8) * 32 + sub * 16 + nloc] = (__bf16)0.f;
      }
    }
    // LDS is in-order within a wave; fence the compiler + outstanding DS ops
    asm volatile("s_wait_dscnt 0x0" ::: "memory");

    // P tile as A-fragment (16 x 32) from LDS
    const v16bf ap = load_a_frag(myp + nloc * 32, 0, lane);

    // V B-fragments: B[k][n] = V[key k][dim n]; strided per-lane gather
    const int kstart = klo + pp * 32;
    const int off2 = (lane >= 16) ? 16 : 0;
#pragma unroll
    for (int vt = 0; vt < 4; ++vt) {
      v16bf bvf;
#pragma unroll
      for (int jj = 0; jj < 16; ++jj) {
        int key = kstart + off2 + jj;
        key = (key < S_) ? key : (S_ - 1);     // clamped; p==0 beyond band
        bvf[jj] = vbase[(size_t)key * DH_ + vt * 16 + nloc];
      }
      acc[vt] = wmma_bf16(ap, bvf, acc[vt]);
    }
    asm volatile("" ::: "memory");  // keep next LDS writes after these reads
  }

  // ctx -> [B, S, H*dh] bf16 (row-major [B*S, D])
#pragma unroll
  for (int vt = 0; vt < 4; ++vt)
#pragma unroll
    for (int r = 0; r < 8; ++r) {
      const int i = q0 + r + half8;
      ctxb[((size_t)bb * S_ + i) * D_ + h * DH_ + vt * 16 + nloc] =
          (__bf16)acc[vt][r];
    }
}

// ---------------------------------------------------------------------------
// Output projection: out = ctx @ Wo^T + bo  (fp32 into d_out)
// Register-blocked 2x4 like qkv_gemm.  grid = (M/32/8, D/64), block = 256
// ---------------------------------------------------------------------------
__global__ __launch_bounds__(256) void out_gemm(
    const __bf16* __restrict__ ctxb, const __bf16* __restrict__ wo,
    const float* __restrict__ bo, float* __restrict__ out) {
  const int lane = threadIdx.x & 31;
  const int wid  = threadIdx.x >> 5;
  const int mt   = blockIdx.x * 8 + wid;
  const int nt   = blockIdx.y;

  const int m0 = mt * (16 * MB_), n0 = nt * (16 * NB_);
  const int nloc = lane & 15;

  const __bf16* arow[MB_];
#pragma unroll
  for (int mi = 0; mi < MB_; ++mi)
    arow[mi] = ctxb + (size_t)(m0 + mi * 16 + nloc) * D_;
  const __bf16* brow[NB_];
#pragma unroll
  for (int ni = 0; ni < NB_; ++ni)
    brow[ni] = wo + (size_t)(n0 + ni * 16 + nloc) * D_;

  v8f acc[MB_][NB_];
#pragma unroll
  for (int mi = 0; mi < MB_; ++mi)
#pragma unroll
    for (int ni = 0; ni < NB_; ++ni) acc[mi][ni] = (v8f){};

#pragma unroll 2
  for (int kb = 0; kb < D_; kb += 32) {
    v16bf a[MB_], b[NB_];
#pragma unroll
    for (int mi = 0; mi < MB_; ++mi) a[mi] = load_a_frag(arow[mi], kb, lane);
#pragma unroll
    for (int ni = 0; ni < NB_; ++ni) b[ni] = load_b_frag_row(brow[ni], kb, lane);
#pragma unroll
    for (int mi = 0; mi < MB_; ++mi)
#pragma unroll
      for (int ni = 0; ni < NB_; ++ni)
        acc[mi][ni] = wmma_bf16(a[mi], b[ni], acc[mi][ni]);
  }

  const int half8 = (lane >= 16) ? 8 : 0;
#pragma unroll
  for (int ni = 0; ni < NB_; ++ni) {
    const int ncol = n0 + ni * 16 + nloc;
    const float bval = bo[ncol];
#pragma unroll
    for (int mi = 0; mi < MB_; ++mi) {
#pragma unroll
      for (int r = 0; r < 8; ++r) {
        const int m = m0 + mi * 16 + r + half8;
        out[(size_t)m * D_ + ncol] = acc[mi][ni][r] + bval;
      }
    }
  }
}

// ---------------------------------------------------------------------------
// Host-side launcher
// ---------------------------------------------------------------------------
extern "C" void kernel_launch(void* const* d_in, const int* in_sizes, int n_in,
                              void* d_out, int out_size, void* d_ws,
                              size_t ws_size, hipStream_t stream) {
  (void)in_sizes; (void)n_in; (void)out_size; (void)ws_size;

  const float* x  = (const float*)d_in[0];
  const float* Wq = (const float*)d_in[1];
  const float* bq = (const float*)d_in[2];
  const float* Wk = (const float*)d_in[3];
  const float* bk = (const float*)d_in[4];
  const float* Wv = (const float*)d_in[5];
  const float* bv = (const float*)d_in[6];
  const float* Wo = (const float*)d_in[7];
  const float* bo = (const float*)d_in[8];

  float* out  = (float*)d_out;                       // [B,S,D]
  float* attn = out + (size_t)B_ * S_ * D_;          // [B,H,S,S]

  // workspace layout (bf16 elements)
  const size_t nX = (size_t)B_ * S_ * D_;            // 4 Mi elems
  const size_t nW = (size_t)D_ * D_;                 // 1 Mi elems
  char* ws = (char*)d_ws;
  __bf16* xb   = (__bf16*)ws;
  __bf16* wqb  = xb  + nX;
  __bf16* wkb  = wqb + nW;
  __bf16* wvb  = wkb + nW;
  __bf16* wob  = wvb + nW;
  __bf16* qb   = wob + nW;
  __bf16* kbuf = qb  + nX;
  __bf16* vbuf = kbuf + nX;
  __bf16* ctxb = vbuf + nX;

  // 1) fp32 -> bf16 converts
  cvt_f32_bf16<<<(int)((nX + 255) / 256), 256, 0, stream>>>(x,  xb,  (int)nX);
  cvt_f32_bf16<<<(int)((nW + 255) / 256), 256, 0, stream>>>(Wq, wqb, (int)nW);
  cvt_f32_bf16<<<(int)((nW + 255) / 256), 256, 0, stream>>>(Wk, wkb, (int)nW);
  cvt_f32_bf16<<<(int)((nW + 255) / 256), 256, 0, stream>>>(Wv, wvb, (int)nW);
  cvt_f32_bf16<<<(int)((nW + 255) / 256), 256, 0, stream>>>(Wo, wob, (int)nW);

  // 2) fused QKV projection (WMMA, 2x4 register-blocked)
  dim3 g1((B_ * S_) / 32 / 8, D_ / 64, 3);
  qkv_gemm<<<g1, 256, 0, stream>>>(xb, wqb, wkb, wvb, bq, bk, bv,
                                   qb, kbuf, vbuf);

  // 3) zero the dense attn output (band is overwritten afterwards)
  const size_t n4 = (size_t)B_ * H_ * S_ * S_ / 4;
  zero_f4<<<(int)((n4 + 255) / 256), 256, 0, stream>>>((float4*)attn, n4);

  // 4) band attention (WMMA scores + WMMA P*V)
  dim3 g2(S_ / 16 / 4, H_, B_);
  attn_kernel<<<g2, 128, 0, stream>>>(qb, kbuf, vbuf, ctxb, attn);

  // 5) output projection (WMMA, 2x4 register-blocked, fp32 out)
  dim3 g3((B_ * S_) / 32 / 8, D_ / 64);
  out_gemm<<<g3, 256, 0, stream>>>(ctxb, wob, bo, out);
}
